// MultiHeadAttention_57037165691092
// MI455X (gfx1250) — matrix operand order repaired and checked by reading, not compiled, not run on hardware
//
#include <hip/hip_runtime.h>

#define EMBED 1024
#define SEQL  2048
#define NHEAD 16
#define HDIM  64
#define BATCH 2
#define MROWS (BATCH*SEQL)   // 4096

typedef __attribute__((ext_vector_type(16))) __bf16 v16bf;
typedef __attribute__((ext_vector_type(8)))  float  v8f;

struct alignas(16) U4 { unsigned int x, y, z, w; };
struct alignas(32) Pack32 { U4 lo, hi; };

__device__ __forceinline__ unsigned short f2bf(float f) {
  unsigned int u = __float_as_uint(f);
  unsigned int r = u + 0x7FFFu + ((u >> 16) & 1u);   // round-to-nearest-even
  return (unsigned short)(r >> 16);
}

// A-matrix tile fragment: 16 bf16 per lane as two 8-element chunks 16 elems apart.
// caller passes p = base + row*ld + (half*8); elems [0..7] = k(half*8+j), [8..15] = k(16+half*8+j)
__device__ __forceinline__ v16bf loadA16(const unsigned short* p) {
  Pack32 s;
  s.lo = *(const U4*)(p);
  s.hi = *(const U4*)(p + 16);
  return __builtin_bit_cast(v16bf, s);
}

// B-matrix tile fragment: two 8-element chunks 16 elems apart; p = base + half*8; elems [0..7] = k(half*8+j), [8..15] = k(16+half*8+j)
__device__ __forceinline__ v16bf loadB16(const unsigned short* p) {
  Pack32 s;
  s.lo = *(const U4*)(p);
  s.hi = *(const U4*)(p + 16);
  return __builtin_bit_cast(v16bf, s);
}

__device__ __forceinline__ v8f wmma_bf16(v16bf a, v16bf b, v8f c) {
  return __builtin_amdgcn_wmma_f32_16x16x32_bf16(false, a, false, b, (short)0, c, false, false);
}

__device__ __forceinline__ v8f vzero8() {
  v8f z;
  #pragma unroll
  for (int i = 0; i < 8; ++i) z[i] = 0.0f;
  return z;
}

// ---------------- conversion kernels ----------------

__global__ void cvt_f32_bf16(const float* __restrict__ in, unsigned short* __restrict__ out, int n) {
  int i = blockIdx.x * blockDim.x + threadIdx.x;
  if (i < n) out[i] = f2bf(in[i]);
}

// W [K,N] fp32 row-major  ->  Wt [N,K] bf16 row-major
__global__ void cvt_w_T(const float* __restrict__ W, unsigned short* __restrict__ Wt) {
  int i = blockIdx.x * blockDim.x + threadIdx.x;
  if (i < EMBED * EMBED) {
    int n = i / EMBED, k = i % EMBED;
    Wt[i] = f2bf(W[k * EMBED + n]);
  }
}

// ---------------- projection GEMM: C = A(bf16,[M,K]) * Bt(bf16,[N,K])^T ----------------
// wave tile 32Mx64N, block = 4 waves stacked in M (128Mx64N). mode 0: out [B,H,S,D]; mode 1: out [B,H,D,S]
__global__ void gemm_qkv(const unsigned short* __restrict__ A,
                         const unsigned short* __restrict__ Bt,
                         unsigned short* __restrict__ dst, int mode) {
  const int lane = threadIdx.x & 31, wave = threadIdx.x >> 5;
  const int half = lane >> 4, l16 = lane & 15;
  const int wm0 = blockIdx.x * 128 + wave * 32;
  const int n0  = blockIdx.y * 64;

  v8f c[2][4];
  #pragma unroll
  for (int i = 0; i < 2; ++i)
    #pragma unroll
    for (int j = 0; j < 4; ++j) c[i][j] = vzero8();

  for (int k0 = 0; k0 < EMBED; k0 += 32) {
    v16bf a[2], b[4];
    #pragma unroll
    for (int i = 0; i < 2; ++i)
      a[i] = loadA16(A + (size_t)(wm0 + i * 16 + l16) * EMBED + k0 + half * 8);
    #pragma unroll
    for (int j = 0; j < 4; ++j)
      b[j] = loadB16(Bt + (size_t)(n0 + j * 16 + l16) * EMBED + k0 + half * 8);
    #pragma unroll
    for (int i = 0; i < 2; ++i)
      #pragma unroll
      for (int j = 0; j < 4; ++j)
        c[i][j] = wmma_bf16(a[i], b[j], c[i][j]);
  }

  #pragma unroll
  for (int i = 0; i < 2; ++i)
    #pragma unroll
    for (int j = 0; j < 4; ++j)
      #pragma unroll
      for (int r = 0; r < 8; ++r) {
        int row = wm0 + i * 16 + half * 8 + r;       // global (b*S + s)
        int col = n0 + j * 16 + l16;                 // embed index
        int bb = row >> 11, ss = row & (SEQL - 1);
        int hh = col >> 6,  dd = col & (HDIM - 1);
        unsigned short v = f2bf(c[i][j][r]);
        if (mode == 0)
          dst[(((size_t)(bb * NHEAD + hh) * SEQL + ss) * HDIM) + dd] = v;
        else
          dst[(((size_t)(bb * NHEAD + hh) * HDIM + dd) * SEQL) + ss] = v;
      }
}

// ---------------- flash attention (causal), 1 wave per 16-query tile ----------------
__global__ void attn_fwd(const unsigned short* __restrict__ Q,   // [B,H,S,D] bf16
                         const unsigned short* __restrict__ K,   // [B,H,S,D] bf16
                         const unsigned short* __restrict__ Vt,  // [B,H,D,S] bf16
                         unsigned short* __restrict__ ctx) {     // [B,S,E] bf16
  __shared__ alignas(16) unsigned short pbuf[4][16 * 32];
  const int lane = threadIdx.x & 31, wave = threadIdx.x >> 5;
  const int half = lane >> 4, l16 = lane & 15;
  const int bh = blockIdx.y;                 // b*NHEAD + h
  const int s0 = (blockIdx.x * 4 + wave) * 16;

  const unsigned short* qb = Q  + (size_t)bh * SEQL * HDIM;
  const unsigned short* kb = K  + (size_t)bh * SEQL * HDIM;
  const unsigned short* vb = Vt + (size_t)bh * HDIM * SEQL;

  v16bf qa0 = loadA16(qb + (size_t)(s0 + l16) * HDIM + half * 8);        // d 0..31
  v16bf qa1 = loadA16(qb + (size_t)(s0 + l16) * HDIM + 32 + half * 8);   // d 32..63

  float mrow[8], lrow[8];
  v8f acc[4];
  #pragma unroll
  for (int r = 0; r < 8; ++r) { mrow[r] = -INFINITY; lrow[r] = 0.0f; }
  #pragma unroll
  for (int t = 0; t < 4; ++t) acc[t] = vzero8();

  unsigned short* pl = pbuf[wave];

  for (int kc0 = 0; kc0 < s0 + 16; kc0 += 32) {
    // ---- scores: two 16x16 tiles, K-dim = head dim (64 = 2 x 32) ----
    v8f sc[2];
    #pragma unroll
    for (int t = 0; t < 2; ++t) {
      int key = kc0 + t * 16 + l16;
      v16bf kb0 = loadB16(kb + (size_t)key * HDIM + half * 8);        // d 0..31
      v16bf kb1 = loadB16(kb + (size_t)key * HDIM + 32 + half * 8);   // d 32..63
      v8f z = vzero8();
      z = wmma_bf16(qa0, kb0, z);
      z = wmma_bf16(qa1, kb1, z);
      sc[t] = z;
    }
    // ---- scale + causal mask (branch-free) ----
    #pragma unroll
    for (int t = 0; t < 2; ++t)
      #pragma unroll
      for (int r = 0; r < 8; ++r) {
        int key  = kc0 + t * 16 + l16;
        int qrow = s0 + half * 8 + r;
        float v = sc[t][r] * 0.125f;                 // 1/sqrt(64)
        sc[t][r] = (key > qrow) ? -INFINITY : v;
      }
    // ---- online softmax; rows of a C tile live across a 16-lane half ----
    #pragma unroll
    for (int r = 0; r < 8; ++r) {
      float mx = fmaxf(sc[0][r], sc[1][r]);
      #pragma unroll
      for (int off = 8; off > 0; off >>= 1) mx = fmaxf(mx, __shfl_xor(mx, off, 16));
      float mnew  = fmaxf(mrow[r], mx);
      float alpha = __expf(mrow[r] - mnew);
      mrow[r] = mnew;
      float p0 = __expf(sc[0][r] - mnew);
      float p1 = __expf(sc[1][r] - mnew);
      float s  = p0 + p1;
      #pragma unroll
      for (int off = 8; off > 0; off >>= 1) s += __shfl_xor(s, off, 16);
      lrow[r] = lrow[r] * alpha + s;
      #pragma unroll
      for (int t = 0; t < 4; ++t) acc[t][r] *= alpha;
      sc[0][r] = p0; sc[1][r] = p1;
    }
    // ---- P: C-layout fp32 -> A-layout bf16 via LDS ----
    #pragma unroll
    for (int t = 0; t < 2; ++t)
      #pragma unroll
      for (int r = 0; r < 8; ++r)
        pl[(half * 8 + r) * 32 + t * 16 + l16] = f2bf(sc[t][r]);
    v16bf pa = loadA16(pl + l16 * 32 + half * 8);
    // ---- ctx += P @ V : 4 N-tiles of 16 dims, K-dim = 32 keys ----
    #pragma unroll
    for (int nt = 0; nt < 4; ++nt) {
      v16bf vv = loadB16(vb + (size_t)(nt * 16 + l16) * SEQL + kc0 + half * 8);
      acc[nt] = wmma_bf16(pa, vv, acc[nt]);
    }
  }

  const int b = bh >> 4, h = bh & (NHEAD - 1);
  #pragma unroll
  for (int nt = 0; nt < 4; ++nt)
    #pragma unroll
    for (int r = 0; r < 8; ++r) {
      int srow = s0 + half * 8 + r;
      int e = h * HDIM + nt * 16 + l16;
      ctx[((size_t)b * SEQL + srow) * EMBED + e] = f2bf(acc[nt][r] / lrow[r]);
    }
}

// ---------------- output projection: out = ctx @ Wo + bias (fp32 out) ----------------
__global__ void gemm_out(const unsigned short* __restrict__ A,   // ctx bf16 [M,E]
                         const unsigned short* __restrict__ Bt,  // Wo^T bf16 [N,K]
                         const float* __restrict__ bias,
                         float* __restrict__ out) {
  const int lane = threadIdx.x & 31, wave = threadIdx.x >> 5;
  const int half = lane >> 4, l16 = lane & 15;
  const int wm0 = blockIdx.x * 128 + wave * 32;
  const int n0  = blockIdx.y * 64;

  v8f c[2][4];
  #pragma unroll
  for (int i = 0; i < 2; ++i)
    #pragma unroll
    for (int j = 0; j < 4; ++j) c[i][j] = vzero8();

  for (int k0 = 0; k0 < EMBED; k0 += 32) {
    v16bf a[2], b[4];
    #pragma unroll
    for (int i = 0; i < 2; ++i)
      a[i] = loadA16(A + (size_t)(wm0 + i * 16 + l16) * EMBED + k0 + half * 8);
    #pragma unroll
    for (int j = 0; j < 4; ++j)
      b[j] = loadB16(Bt + (size_t)(n0 + j * 16 + l16) * EMBED + k0 + half * 8);
    #pragma unroll
    for (int i = 0; i < 2; ++i)
      #pragma unroll
      for (int j = 0; j < 4; ++j)
        c[i][j] = wmma_bf16(a[i], b[j], c[i][j]);
  }

  #pragma unroll
  for (int i = 0; i < 2; ++i)
    #pragma unroll
    for (int j = 0; j < 4; ++j)
      #pragma unroll
      for (int r = 0; r < 8; ++r) {
        int row = wm0 + i * 16 + half * 8 + r;
        int col = n0 + j * 16 + l16;
        out[(size_t)row * EMBED + col] = c[i][j][r] + bias[col];
      }
}

extern "C" void kernel_launch(void* const* d_in, const int* in_sizes, int n_in,
                              void* d_out, int out_size, void* d_ws, size_t ws_size,
                              hipStream_t stream) {
  const float* x   = (const float*)d_in[0];
  const float* W_q = (const float*)d_in[1];
  const float* W_k = (const float*)d_in[2];
  const float* W_v = (const float*)d_in[3];
  const float* W_o = (const float*)d_in[4];
  const float* b_o = (const float*)d_in[5];
  float* out = (float*)d_out;

  char* w = (char*)d_ws;
  const size_t SZ_X = (size_t)MROWS * EMBED * 2;        // 8 MB
  const size_t SZ_W = (size_t)EMBED * EMBED * 2;        // 2 MB
  unsigned short* xbf  = (unsigned short*)(w);
  unsigned short* wqt  = (unsigned short*)(w + SZ_X);
  unsigned short* wkt  = (unsigned short*)(w + SZ_X + SZ_W);
  unsigned short* wvt  = (unsigned short*)(w + SZ_X + 2 * SZ_W);
  unsigned short* wot  = (unsigned short*)(w + SZ_X + 3 * SZ_W);
  unsigned short* qb   = (unsigned short*)(w + SZ_X + 4 * SZ_W);
  unsigned short* kb   = (unsigned short*)(w + 2 * SZ_X + 4 * SZ_W);
  unsigned short* vb   = (unsigned short*)(w + 3 * SZ_X + 4 * SZ_W);
  unsigned short* ctxb = (unsigned short*)(w + 4 * SZ_X + 4 * SZ_W);

  int nx = MROWS * EMBED;
  cvt_f32_bf16<<<(nx + 255) / 256, 256, 0, stream>>>(x, xbf, nx);
  int nw = EMBED * EMBED;
  cvt_w_T<<<(nw + 255) / 256, 256, 0, stream>>>(W_q, wqt);
  cvt_w_T<<<(nw + 255) / 256, 256, 0, stream>>>(W_k, wkt);
  cvt_w_T<<<(nw + 255) / 256, 256, 0, stream>>>(W_v, wvt);
  cvt_w_T<<<(nw + 255) / 256, 256, 0, stream>>>(W_o, wot);

  dim3 gg(MROWS / 128, EMBED / 64);   // 32 x 16
  gemm_qkv<<<gg, 128, 0, stream>>>(xbf, wqt, qb, 0);
  gemm_qkv<<<gg, 128, 0, stream>>>(xbf, wkt, kb, 0);
  gemm_qkv<<<gg, 128, 0, stream>>>(xbf, wvt, vb, 1);

  attn_fwd<<<dim3(SEQL / 64, BATCH * NHEAD), 128, 0, stream>>>(qb, kb, vb, ctxb);

  gemm_out<<<gg, 128, 0, stream>>>(ctxb, wot, b_o, out);
}
